// TMM_81441169866922
// MI455X (gfx1250) — matrix operand order repaired and checked by reading, not compiled, not run on hardware
//
#include <hip/hip_runtime.h>
#include <hip/hip_bf16.h>
#include <math.h>

// ---------------- problem constants ----------------
#define B_    8
#define L_    2048
#define DIM_  128
#define DH_   64
#define RANK_ 32
#define NS_   64
#define R_    (B_ * L_)      // 16384 rows
#define XPJ_  160            // RANK + 2*NS
#define EPS_  1e-5f

typedef _Float16 f16;
typedef __attribute__((ext_vector_type(16))) _Float16 v16h;
typedef __attribute__((ext_vector_type(8)))  _Float16 v8h;
typedef __attribute__((ext_vector_type(8)))  float    v8f;

__device__ __forceinline__ float sigmoidf_(float x) { return 1.0f / (1.0f + __expf(-x)); }
__device__ __forceinline__ float siluf_(float x)    { return x * sigmoidf_(x); }

// ---------------- workspace layout (bytes, all 256-aligned) ----------------
#define OFF_X1B   0u                     // R*DH f32   (x @ w1 + b1)
#define OFF_X2B   4194304u               // R*DH f32   (x @ w2 + b2)
#define OFF_UP    8388608u               // R*DH f32   u' = silu(conv) + t_emb*scale
#define OFF_X2C   12582912u              // R*DH f32   silu(dilconv)
#define OFF_XPJ   16777216u              // R*160 f32  x_proj (delta|B'|C')
#define OFF_DLT   27262976u              // R*DH f32   delta' (softplus)
#define OFF_UD    31457280u              // R*DH f32   u' * delta'
#define OFF_V     35651584u              // R*DH f32   scan output v
#define OFF_DAF   39845888u              // R*4096 f16 dA in WMMA-A fragment order

// ============================================================================
// K1: x @ w1 + b1 and x @ w2 + b2 (shared x loads)
// ============================================================================
__global__ __launch_bounds__(256) void k_proj(
    const float* __restrict__ x, const float* __restrict__ w1, const float* __restrict__ b1,
    const float* __restrict__ w2, const float* __restrict__ b2,
    float* __restrict__ x1b, float* __restrict__ x2b) {
  int t = blockIdx.x * blockDim.x + threadIdx.x;   // R_*DH_ threads
  int r = t >> 6, o = t & 63;
  const float* xr = x + (size_t)r * DIM_;
  float a = b1[o], c = b2[o];
#pragma unroll 4
  for (int d = 0; d < DIM_; ++d) {
    float xv = xr[d];
    a += xv * w1[d * DH_ + o];
    c += xv * w2[d * DH_ + o];
  }
  x1b[t] = a;
  x2b[t] = c;
}

// ============================================================================
// K2: two channel-mixing conv1d (k=3; pad1/dil1 and pad2/dil2) + SiLU,
//     then u' = branch1 + t_emb*scale
// ============================================================================
__global__ __launch_bounds__(256) void k_conv(
    const float* __restrict__ x1b, const float* __restrict__ x2b,
    const float* __restrict__ wdep, const float* __restrict__ bdep,
    const float* __restrict__ wdil, const float* __restrict__ bdil,
    const float* __restrict__ t_emb, const float* __restrict__ scale,
    float* __restrict__ up, float* __restrict__ x2c) {
  int t = blockIdx.x * blockDim.x + threadIdx.x;   // R_*DH_ threads
  int r = t >> 6, o = t & 63;
  int b = r / L_, l = r % L_;
  float s1 = bdep[o], s2 = bdil[o];
#pragma unroll
  for (int tap = 0; tap < 3; ++tap) {
    int l1 = l - 1 + tap;         // pad=1, dil=1
    int l2 = l - 2 + 2 * tap;     // pad=2, dil=2
    const float* w1p = wdep + o * DH_ * 3 + tap;   // [o][i][tap], stride 3 over i
    const float* w2p = wdil + o * DH_ * 3 + tap;
    if (l1 >= 0 && l1 < L_) {
      const float* xp = x1b + ((size_t)(b * L_ + l1)) * DH_;
      for (int i = 0; i < DH_; ++i) s1 += w1p[i * 3] * xp[i];
    }
    if (l2 >= 0 && l2 < L_) {
      const float* xp = x2b + ((size_t)(b * L_ + l2)) * DH_;
      for (int i = 0; i < DH_; ++i) s2 += w2p[i * 3] * xp[i];
    }
  }
  up[t]  = siluf_(s1) + t_emb[b * DH_ + o] * scale[0];
  x2c[t] = siluf_(s2);
}

// ============================================================================
// K3: x_proj = u' @ w_in + b_in     (one row per block, 160 threads)
// ============================================================================
__global__ __launch_bounds__(XPJ_) void k_xproj(
    const float* __restrict__ up, const float* __restrict__ w_in,
    const float* __restrict__ b_in, float* __restrict__ xproj) {
  int r = blockIdx.x, j = threadIdx.x;
  __shared__ float s_u[DH_];
  if (j < DH_) s_u[j] = up[(size_t)r * DH_ + j];
  __syncthreads();
  float s = b_in[j];
#pragma unroll 4
  for (int d = 0; d < DH_; ++d) s += s_u[d] * w_in[d * XPJ_ + j];
  xproj[(size_t)r * XPJ_ + j] = s;
}

// ============================================================================
// K4: delta' = softplus(delta @ w_dl + b_dl);  ud = u' * delta'
// ============================================================================
__global__ __launch_bounds__(256) void k_delta(
    const float* __restrict__ xproj, const float* __restrict__ w_dl,
    const float* __restrict__ b_dl, const float* __restrict__ up,
    float* __restrict__ deltap, float* __restrict__ ud) {
  int t = blockIdx.x * blockDim.x + threadIdx.x;   // R_*DH_ threads
  int r = t >> 6, d = t & 63;
  const float* xp = xproj + (size_t)r * XPJ_;
  float s = b_dl[d];
#pragma unroll 4
  for (int k = 0; k < RANK_; ++k) s += xp[k] * w_dl[k * DH_ + d];
  float dp = (s > 20.0f) ? s : log1pf(__expf(s));
  deltap[t] = dp;
  ud[t]     = up[t] * dp;
}

// ============================================================================
// K5: pack dA = exp(delta'[d] * A[d,k]) as f16 in WMMA A-fragment order.
//   fragment f = ti*2 + kc (ti = 16-row tile of d, kc = 32-wide chunk of k)
//   lane < 16: d = ti*16+lane,    k = kc*32 + (e<8 ? e   : e+8)
//   lane >=16: d = ti*16+lane-16, k = kc*32 + (e<8 ? e+8 : e+16)
//   linear f16 index = r*4096 + f*512 + lane*16 + e
// ============================================================================
__global__ __launch_bounds__(256) void k_packA(
    const float* __restrict__ deltap, const float* __restrict__ A,
    f16* __restrict__ dAf) {
  int tid = blockIdx.x * blockDim.x + threadIdx.x;   // R_*2048 threads (2 elems each)
  int r    = tid >> 11;
  int rem  = tid & 2047;
  int f    = rem >> 8;
  int lane = (rem >> 3) & 31;
  int j    = rem & 7;
  int ti = f >> 1, kc = f & 1;
  int d  = ti * 16 + (lane & 15);
  int e0 = 2 * j;                       // e0, e0+1 are in the same <8 / >=8 half
  int k0;
  if (lane < 16) k0 = kc * 32 + ((e0 < 8) ? e0 : e0 + 8);
  else           k0 = kc * 32 + ((e0 < 8) ? e0 + 8 : e0 + 16);
  float dpv = deltap[(size_t)r * DH_ + d];
  const float* Ar = A + d * NS_;
  f16 h0 = (f16)__expf(dpv * Ar[k0]);
  f16 h1 = (f16)__expf(dpv * Ar[k0 + 1]);
  size_t base = (size_t)tid * 2;
  dAf[base]     = h0;
  dAf[base + 1] = h1;
}

// ============================================================================
// K6: the sequential scan. One workgroup (8 wave32) per batch.
//   Z_l = DA_l @ Z_{l-1} + ud_l (x) B'_l   (64x64, f16 operands, f32 acc, WMMA)
//   v_l = (Z_l @ C'_l) * sigmoid(t_emb)
//   Z kept in LDS transposed [n][k], rows padded to 72 f16 for aligned v8h IO.
//   Next step's dA fragments (8KB) and ud/B'/C' (768B) are prefetched one step
//   ahead via GLOBAL_LOAD_ASYNC_TO_LDS (ASYNCcnt), double-buffered in LDS, and
//   retired with s_wait_asynccnt + barrier — hiding global latency behind WMMA.
// ============================================================================
__global__ __launch_bounds__(256, 1) void k_scan(
    const f16* __restrict__ dAf, const float* __restrict__ ud,
    const float* __restrict__ xproj, const float* __restrict__ t_emb,
    float* __restrict__ v) {
  const int b    = blockIdx.x;
  const int t    = threadIdx.x;
  const int lane = t & 31;
  const int w    = t >> 5;
  const int ti   = w & 3;            // output row tile (d)
  const int tj0  = (w >> 2) * 2;     // two output col tiles (n)

  __shared__ __align__(32) f16   s_A[2][4096];   // double-buffered dA fragments
  __shared__ __align__(16) float s_vec[2][192];  // [ ud(64) | B'(64) | C'(64) ]
  __shared__ __align__(32) f16   s_Z[64 * 72];   // Z^T: [n][k], padded rows
  __shared__ float s_sig[DH_], s_part[256];

  // async prefetch of one step's operands into LDS buffer `buf`
#define PREFETCH_STEP(lidx, buf)                                                 \
  {                                                                              \
    const size_t rr = (size_t)b * L_ + (size_t)(lidx);                           \
    const f16* ga   = dAf + rr * 4096 + (size_t)t * 16;   /* 32 B per thread */  \
    unsigned   la   = (unsigned)(uintptr_t)(&s_A[(buf)][t * 16]);                \
    asm volatile("global_load_async_to_lds_b128 %0, %1, off"                     \
                 :: "v"(la), "v"(ga) : "memory");                                \
    asm volatile("global_load_async_to_lds_b128 %0, %1, off offset:16"          \
                 :: "v"(la), "v"(ga) : "memory");                                \
    if (t < 192) {                                                               \
      const float* gs;                                                           \
      if (t < 64)       gs = ud + rr * DH_ + t;                                  \
      else if (t < 128) gs = xproj + rr * XPJ_ + RANK_ + (t - 64);               \
      else              gs = xproj + rr * XPJ_ + RANK_ + NS_ + (t - 128);        \
      unsigned lv = (unsigned)(uintptr_t)(&s_vec[(buf)][t]);                     \
      asm volatile("global_load_async_to_lds_b32 %0, %1, off"                    \
                   :: "v"(lv), "v"(gs) : "memory");                              \
    }                                                                            \
  }

  if (t < DH_) s_sig[t] = sigmoidf_(t_emb[b * DH_ + t]);
  for (int i = t; i < 64 * 72; i += 256) s_Z[i] = (f16)0.0f;

  PREFETCH_STEP(0, 0);
  asm volatile("s_wait_asynccnt 0x0" ::: "memory");
  __syncthreads();

  const int nloc  = lane & 15;
  const int mbase = (lane < 16) ? 0 : 8;
  const int k0a   = (lane < 16) ? 0 : 8;    // B-fragment k base within 32-chunk

  for (int l = 0; l < L_; ++l) {
    const int cur = l & 1;
    const size_t r = (size_t)b * L_ + l;

    // ---- kick off next step's DMA while this step computes
    if (l + 1 < L_) PREFETCH_STEP(l + 1, cur ^ 1);

    // ---- A fragments from LDS staging (pre-swizzled, 32B contiguous per lane)
    const f16* ab = &s_A[cur][ti * 1024 + lane * 16];
    v16h a0 = *(const v16h*)(ab);
    v16h a1 = *(const v16h*)(ab + 512);

    const float* vud = &s_vec[cur][0];
    const float* vbp = &s_vec[cur][64];
    const float* vcp = &s_vec[cur][128];

    v8f acc[2];
#pragma unroll
    for (int q = 0; q < 2; ++q) {
      const int tj = tj0 + q;
      const int n  = tj * 16 + nloc;
      // C init = rank-1 dBu: ud[d] * B'[n]
      v8f c;
      const float bpn = vbp[n];
#pragma unroll
      for (int i = 0; i < 8; ++i) c[i] = vud[ti * 16 + mbase + i] * bpn;
      // B fragments from Z^T (two runs of 8 f16 per lane per k-chunk, at k0a and k0a+16)
      const f16* zr = &s_Z[n * 72];
      union { v16h v; v8h h[2]; } bf0, bf1;
      bf0.h[0] = *(const v8h*)(zr + k0a);
      bf0.h[1] = *(const v8h*)(zr + k0a + 16);
      bf1.h[0] = *(const v8h*)(zr + 32 + k0a);
      bf1.h[1] = *(const v8h*)(zr + 32 + k0a + 16);
      c = __builtin_amdgcn_wmma_f32_16x16x32_f16(false, a0, false, bf0.v,
                                                 (short)0, c, false, false);
      c = __builtin_amdgcn_wmma_f32_16x16x32_f16(false, a1, false, bf1.v,
                                                 (short)0, c, false, false);
      acc[q] = c;
    }
    __syncthreads();   // all Z reads complete before overwrite

    // ---- write new Z^T (8 consecutive f16 per lane per tile)
#pragma unroll
    for (int q = 0; q < 2; ++q) {
      const int n = (tj0 + q) * 16 + nloc;
      f16* zw = &s_Z[n * 72 + ti * 16 + mbase];
#pragma unroll
      for (int i = 0; i < 8; ++i) zw[i] = (f16)acc[q][i];
    }
    __syncthreads();

    // ---- v_l[d] = sigmoid(t)[d] * sum_n Z[d,n] * C'[n]
    {
      const int d = t & 63, g = t >> 6;
      float p = 0.0f;
#pragma unroll
      for (int jj = 0; jj < 16; ++jj) {
        const int n = g * 16 + jj;
        p += (float)s_Z[n * 72 + d] * vcp[n];
      }
      s_part[t] = p;
      __syncthreads();
      if (t < 64) {
        float s = (s_part[t] + s_part[t + 64] + s_part[t + 128] + s_part[t + 192]) * s_sig[t];
        v[r * DH_ + t] = s;
      }
    }

    // ---- retire the prefetch for step l+1, publish the staging buffers
    asm volatile("s_wait_asynccnt 0x0" ::: "memory");
    __syncthreads();
  }
#undef PREFETCH_STEP
}

// ============================================================================
// K7: concat([v + u'*D, x2c]) @ w_out + b_out, then LayerNorm over DIM
// ============================================================================
__global__ __launch_bounds__(DIM_) void k_out(
    const float* __restrict__ v, const float* __restrict__ up,
    const float* __restrict__ x2c, const float* __restrict__ Dp,
    const float* __restrict__ w_out, const float* __restrict__ b_out,
    const float* __restrict__ ln_g, const float* __restrict__ ln_b,
    float* __restrict__ out) {
  int r = blockIdx.x, j = threadIdx.x;   // 128 threads
  __shared__ float s_in[DIM_];
  __shared__ float s_red[DIM_];
  if (j < DH_) s_in[j] = v[(size_t)r * DH_ + j] + up[(size_t)r * DH_ + j] * Dp[j];
  else         s_in[j] = x2c[(size_t)r * DH_ + (j - DH_)];
  __syncthreads();
  float y = b_out[j];
#pragma unroll 4
  for (int i = 0; i < DIM_; ++i) y += s_in[i] * w_out[i * DIM_ + j];
  // mean
  s_red[j] = y; __syncthreads();
  for (int st = 64; st > 0; st >>= 1) { if (j < st) s_red[j] += s_red[j + st]; __syncthreads(); }
  float mu = s_red[0] * (1.0f / DIM_);
  __syncthreads();
  // variance (population)
  float dy = y - mu;
  s_red[j] = dy * dy; __syncthreads();
  for (int st = 64; st > 0; st >>= 1) { if (j < st) s_red[j] += s_red[j + st]; __syncthreads(); }
  float var = s_red[0] * (1.0f / DIM_);
  out[(size_t)r * DIM_ + j] = dy * rsqrtf(var + EPS_) * ln_g[j] + ln_b[j];
}

// ============================================================================
extern "C" void kernel_launch(void* const* d_in, const int* in_sizes, int n_in,
                              void* d_out, int out_size, void* d_ws, size_t ws_size,
                              hipStream_t stream) {
  const float* x     = (const float*)d_in[0];
  const float* t_emb = (const float*)d_in[1];
  const float* w1    = (const float*)d_in[2];
  const float* b1    = (const float*)d_in[3];
  const float* w2    = (const float*)d_in[4];
  const float* b2    = (const float*)d_in[5];
  const float* wdep  = (const float*)d_in[6];
  const float* bdep  = (const float*)d_in[7];
  const float* wdil  = (const float*)d_in[8];
  const float* bdil  = (const float*)d_in[9];
  const float* w_in  = (const float*)d_in[10];
  const float* b_in  = (const float*)d_in[11];
  const float* w_dl  = (const float*)d_in[12];
  const float* b_dl  = (const float*)d_in[13];
  const float* A     = (const float*)d_in[14];
  const float* Dp    = (const float*)d_in[15];
  const float* scale = (const float*)d_in[16];
  const float* w_out = (const float*)d_in[17];
  const float* b_out = (const float*)d_in[18];
  const float* ln_g  = (const float*)d_in[19];
  const float* ln_b  = (const float*)d_in[20];

  char* ws = (char*)d_ws;
  float* x1b    = (float*)(ws + OFF_X1B);
  float* x2b    = (float*)(ws + OFF_X2B);
  float* up     = (float*)(ws + OFF_UP);
  float* x2c    = (float*)(ws + OFF_X2C);
  float* xproj  = (float*)(ws + OFF_XPJ);
  float* deltap = (float*)(ws + OFF_DLT);
  float* udp    = (float*)(ws + OFF_UD);
  float* vv     = (float*)(ws + OFF_V);
  f16*   dAf    = (f16*)  (ws + OFF_DAF);

  float* out = (float*)d_out;

  k_proj <<<(R_ * DH_) / 256, 256, 0, stream>>>(x, w1, b1, w2, b2, x1b, x2b);
  k_conv <<<(R_ * DH_) / 256, 256, 0, stream>>>(x1b, x2b, wdep, bdep, wdil, bdil,
                                                t_emb, scale, up, x2c);
  k_xproj<<<R_, XPJ_, 0, stream>>>(up, w_in, b_in, xproj);
  k_delta<<<(R_ * DH_) / 256, 256, 0, stream>>>(xproj, w_dl, b_dl, up, deltap, udp);
  k_packA<<<(R_ * 2048) / 256, 256, 0, stream>>>(deltap, A, dAf);
  k_scan <<<B_, 256, 0, stream>>>(dAf, udp, xproj, t_emb, vv);
  k_out  <<<R_, DIM_, 0, stream>>>(vv, up, x2c, Dp, w_out, b_out, ln_g, ln_b, out);
}
